// AttentionGuidedEgoGAT_438086664202
// MI455X (gfx1250) — matrix operand order repaired and checked by reading, not compiled, not run on hardware
//
#include <hip/hip_runtime.h>

typedef __attribute__((ext_vector_type(16))) _Float16 v16h;
typedef __attribute__((ext_vector_type(8)))  _Float16 v8h;
typedef __attribute__((ext_vector_type(8)))  float    v8f;

#define THREADS 256

__device__ __forceinline__ float gelu_f(float v) {
  return 0.5f * v * (1.0f + erff(v * 0.70710678118654752440f));
}

// Monotone float atomic-max: signed-int max for non-negatives, unsigned-int min
// for negatives. Both moves are monotone toward the float max. Init pattern
// must be 0xFFFFFFFF (loses to everything on both paths; every node has a
// self-loop so it is always overwritten before being read as a float).
__device__ __forceinline__ void atomicMaxF(float* addr, float v) {
  if (v >= 0.0f) atomicMax((int*)addr, __float_as_int(v));
  else           atomicMin((unsigned int*)addr, __float_as_uint(v));
}

__device__ __forceinline__ void edge_sd(const int* __restrict__ ei, int E, int e,
                                        int& s, int& d) {
  if (e < E) { s = ei[e]; d = ei[E + e]; }   // edge_index rows [0]=src [1]=dst
  else       { s = e - E; d = s; }           // appended self-loops
}

// ---------------------------------------------------------------- utilities
__global__ void fill_u32_k(unsigned int* __restrict__ p, unsigned int v, int n) {
  int i = blockIdx.x * blockDim.x + threadIdx.x;
  if (i < n) p[i] = v;
}

__global__ void f2h_k(const float* __restrict__ in, _Float16* __restrict__ out, int n) {
  int i = blockIdx.x * blockDim.x + threadIdx.x;
  if (i < n) out[i] = (_Float16)in[i];
}

template <int D>
__global__ void addbias_k(float* __restrict__ xv, const float* __restrict__ b, int n) {
  int i = blockIdx.x * blockDim.x + threadIdx.x;
  if (i < n) xv[i] += b[i & (D - 1)];
}

// ---------------------------------------------------------------- WMMA GEMM
// C[M,Nc] = A[M,K](f16) @ B[K,Nc](f16) + bias, f32 accumulate.
// One wave computes a 16 x (NT*16) tile. K is a compile-time constant so the
// k-loop fully unrolls: all A fragment loads become base+immediate-offset
// b128 loads the scheduler can hoist across the WMMA chain (no per-iteration
// address math, no full loadcnt fence per WMMA).
template <int NT, int K>
__global__ __launch_bounds__(THREADS)
void gemm_wmma_k(const _Float16* __restrict__ A, const _Float16* __restrict__ B,
                 const float* __restrict__ bias, float* __restrict__ C,
                 int M, int Nc) {
  const int wave   = blockIdx.x * (THREADS / 32) + (threadIdx.x >> 5);
  const int lane   = threadIdx.x & 31;
  const int tilesN = Nc / (NT * 16);
  const int m0     = (wave / tilesN) * 16;
  if (m0 >= M) return;                       // uniform per wave
  const int n0 = (wave % tilesN) * (NT * 16);

  v8f acc[NT];
#pragma unroll
  for (int t = 0; t < NT; ++t)
#pragma unroll
    for (int j = 0; j < 8; ++j) acc[t][j] = 0.0f;

  int arow = m0 + (lane & 15);
  if (arow >= M) arow = M - 1;
  const int kbase = (lane >> 4) * 8;         // lanes 16-31 hold K+8 block

  const _Float16* abase = A + (size_t)arow * K + kbase;
  const _Float16* bbase = B + (size_t)kbase * Nc + n0 + (lane & 15);   // lane = N column; K rows kbase+[0,8), 16+kbase+[0,8)

#pragma unroll
  for (int k0 = 0; k0 < K; k0 += 32) {
    // A fragment (16x32 f16): lane holds K = k0+kbase+[0,8) and k0+16+kbase+[0,8)
    v8h lo = *reinterpret_cast<const v8h*>(abase + k0);
    v8h hi = *reinterpret_cast<const v8h*>(abase + k0 + 16);
    v16h a;
#pragma unroll
    for (int i = 0; i < 8; ++i) { a[i] = lo[i]; a[i + 8] = hi[i]; }

#pragma unroll
    for (int t = 0; t < NT; ++t) {
      v16h b;
#pragma unroll
      for (int i = 0; i < 8; ++i) {
        b[i]     = bbase[(size_t)(k0 + i) * Nc + t * 16];
        b[i + 8] = bbase[(size_t)(k0 + 16 + i) * Nc + t * 16];
      }
      acc[t] = __builtin_amdgcn_wmma_f32_16x16x32_f16(false, a, false, b,
                                                      (short)0, acc[t], false, false);
    }
  }

  const int col_l = lane & 15;
  const int radd  = (lane >> 4) * 8;         // C layout: VGPR r -> row r (+8)
#pragma unroll
  for (int t = 0; t < NT; ++t) {
    int col = n0 + t * 16 + col_l;
    float bv = bias[col];
#pragma unroll
    for (int r = 0; r < 8; ++r) {
      int row = m0 + radd + r;
      if (row < M) C[(size_t)row * Nc + col] = acc[t][r] + bv;
    }
  }
}

// ------------------------------------------------- GATv2 conv1 (H=4, C=32)
// One wave per edge; lane covers 4 contiguous channels; head = lane>>3.
__global__ __launch_bounds__(THREADS)
void gat1_logit_k(const float* __restrict__ xl, const float* __restrict__ xr,
                  const int* __restrict__ ei, const float* __restrict__ att,
                  float* __restrict__ lg, float* __restrict__ m, int E, int EL) {
  int e = blockIdx.x * 8 + (threadIdx.x >> 5);
  if (e >= EL) return;
  int lane = threadIdx.x & 31;
  int s, d; edge_sd(ei, E, e, s, d);
  float4 a  = reinterpret_cast<const float4*>(xl + (size_t)s * 128)[lane];
  float4 b  = reinterpret_cast<const float4*>(xr + (size_t)d * 128)[lane];
  float4 at = reinterpret_cast<const float4*>(att)[lane];
  float acc = 0.0f, v;
  v = a.x + b.x; v = v > 0.0f ? v : 0.2f * v; acc += v * at.x;
  v = a.y + b.y; v = v > 0.0f ? v : 0.2f * v; acc += v * at.y;
  v = a.z + b.z; v = v > 0.0f ? v : 0.2f * v; acc += v * at.z;
  v = a.w + b.w; v = v > 0.0f ? v : 0.2f * v; acc += v * at.w;
  acc += __shfl_xor(acc, 1, 32);
  acc += __shfl_xor(acc, 2, 32);
  acc += __shfl_xor(acc, 4, 32);
  if ((lane & 7) == 0) {
    int h = lane >> 3;
    lg[(size_t)e * 4 + h] = acc;
    atomicMaxF(m + (size_t)d * 4 + h, acc);
  }
}

__global__ void gat1_exp_k(float* __restrict__ lg, const float* __restrict__ m,
                           float* __restrict__ den, const int* __restrict__ ei,
                           int E, int EL) {
  int idx = blockIdx.x * blockDim.x + threadIdx.x;
  if (idx >= EL * 4) return;
  int e = idx >> 2, h = idx & 3;
  int s, d; edge_sd(ei, E, e, s, d); (void)s;
  float ex = expf(lg[idx] - m[(size_t)d * 4 + h]);
  lg[idx] = ex;
  atomicAdd(&den[(size_t)d * 4 + h], ex);
}

__global__ __launch_bounds__(THREADS)
void gat1_scatter_k(const float* __restrict__ xl, const float* __restrict__ lg,
                    const float* __restrict__ den, const int* __restrict__ ei,
                    float* __restrict__ outp, int E, int EL) {
  int e = blockIdx.x * 8 + (threadIdx.x >> 5);
  if (e >= EL) return;
  int lane = threadIdx.x & 31;
  int s, d; edge_sd(ei, E, e, s, d);
  int h = lane >> 3;
  float alpha = lg[(size_t)e * 4 + h] / den[(size_t)d * 4 + h];
  float4 a = reinterpret_cast<const float4*>(xl + (size_t)s * 128)[lane];
  float* o = outp + (size_t)d * 128 + lane * 4;
  atomicAdd(o + 0, alpha * a.x);
  atomicAdd(o + 1, alpha * a.y);
  atomicAdd(o + 2, alpha * a.z);
  atomicAdd(o + 3, alpha * a.w);
}

// ------------------------------------------------- GATv2 conv2 (H=1, C=32)
__global__ __launch_bounds__(THREADS)
void gat2_logit_k(const float* __restrict__ xl, const float* __restrict__ xr,
                  const int* __restrict__ ei, const float* __restrict__ att,
                  float* __restrict__ lg, float* __restrict__ m, int E, int EL) {
  int e = blockIdx.x * 8 + (threadIdx.x >> 5);
  if (e >= EL) return;
  int lane = threadIdx.x & 31;
  int s, d; edge_sd(ei, E, e, s, d);
  float v = xl[(size_t)s * 32 + lane] + xr[(size_t)d * 32 + lane];
  v = v > 0.0f ? v : 0.2f * v;
  float acc = v * att[lane];
  acc += __shfl_xor(acc, 1, 32);
  acc += __shfl_xor(acc, 2, 32);
  acc += __shfl_xor(acc, 4, 32);
  acc += __shfl_xor(acc, 8, 32);
  acc += __shfl_xor(acc, 16, 32);
  if (lane == 0) {
    lg[e] = acc;
    atomicMaxF(&m[d], acc);
  }
}

__global__ void gat2_exp_k(float* __restrict__ lg, const float* __restrict__ m,
                           float* __restrict__ den, const int* __restrict__ ei,
                           int E, int EL) {
  int e = blockIdx.x * blockDim.x + threadIdx.x;
  if (e >= EL) return;
  int s, d; edge_sd(ei, E, e, s, d); (void)s;
  float ex = expf(lg[e] - m[d]);
  lg[e] = ex;
  atomicAdd(&den[d], ex);
}

__global__ __launch_bounds__(THREADS)
void gat2_scatter_k(const float* __restrict__ xl, const float* __restrict__ lg,
                    const float* __restrict__ den, const int* __restrict__ ei,
                    float* __restrict__ outp, int E, int EL) {
  int e = blockIdx.x * 8 + (threadIdx.x >> 5);
  if (e >= EL) return;
  int lane = threadIdx.x & 31;
  int s, d; edge_sd(ei, E, e, s, d);
  float alpha = lg[e] / den[d];
  atomicAdd(&outp[(size_t)d * 32 + lane], alpha * xl[(size_t)s * 32 + lane]);
}

// ---------------------------------------------------------------- GraphNorm
__global__ void count_k(const int* __restrict__ batch, float* __restrict__ cnt, int N) {
  int n = blockIdx.x * blockDim.x + threadIdx.x;
  if (n < N) atomicAdd(&cnt[batch[n]], 1.0f);
}

template <int D>
__global__ void gn_sum_k(const float* __restrict__ xv, const int* __restrict__ batch,
                         float* __restrict__ s, int total) {
  int i = blockIdx.x * blockDim.x + threadIdx.x;
  if (i >= total) return;
  int n = i / D, c = i & (D - 1);
  atomicAdd(&s[(size_t)batch[n] * D + c], xv[i]);
}

template <int D>
__global__ void gn_mean_k(float* __restrict__ s, const float* __restrict__ cnt,
                          int total) {
  int i = blockIdx.x * blockDim.x + threadIdx.x;
  if (i >= total) return;
  s[i] /= fmaxf(cnt[i / D], 1.0f);
}

template <int D>
__global__ void gn_center_k(float* __restrict__ xv, const int* __restrict__ batch,
                            const float* __restrict__ mean, const float* __restrict__ ms,
                            float* __restrict__ vs, int total) {
  int i = blockIdx.x * blockDim.x + threadIdx.x;
  if (i >= total) return;
  int n = i / D, c = i & (D - 1);
  float o = xv[i] - mean[(size_t)batch[n] * D + c] * ms[c];
  xv[i] = o;
  atomicAdd(&vs[(size_t)batch[n] * D + c], o * o);
}

template <int D>
__global__ void gn_rstd_k(const float* __restrict__ vs, const float* __restrict__ cnt,
                          float* __restrict__ rstd, int total) {
  int i = blockIdx.x * blockDim.x + threadIdx.x;
  if (i >= total) return;
  float var = vs[i] / fmaxf(cnt[i / D], 1.0f);
  rstd[i] = 1.0f / sqrtf(var + 1e-5f);
}

template <int D>
__global__ void gn_fin_k(const float* __restrict__ xv, const int* __restrict__ batch,
                         const float* __restrict__ rstd, const float* __restrict__ w,
                         const float* __restrict__ b, const float* __restrict__ res,
                         float* __restrict__ h, int total) {
  int i = blockIdx.x * blockDim.x + threadIdx.x;
  if (i >= total) return;
  int n = i / D, c = i & (D - 1);
  float y = w[c] * xv[i] * rstd[(size_t)batch[n] * D + c] + b[c];
  h[i] = gelu_f(y + res[i]);
}

// ---------------------------------------------------------------- pooling + head
__global__ void pool_acc_k(const float* __restrict__ h2, const int* __restrict__ batch,
                           float* __restrict__ pm, float* __restrict__ px, int total) {
  int i = blockIdx.x * blockDim.x + threadIdx.x;
  if (i >= total) return;
  int n = i >> 5, c = i & 31;
  int g = batch[n];
  float v = h2[i];
  atomicAdd(&pm[(size_t)g * 32 + c], v);
  atomicMaxF(&px[(size_t)g * 32 + c], v);
}

__global__ void pool_fin_k(const float* __restrict__ pm, float* __restrict__ px,
                           const float* __restrict__ cnt, float* __restrict__ xmean,
                           int total) {
  int i = blockIdx.x * blockDim.x + threadIdx.x;
  if (i >= total) return;
  float cg = cnt[i >> 5];
  xmean[i] = pm[i] / fmaxf(cg, 1.0f);
  px[i] = (cg > 0.0f) ? px[i] : 0.0f;
}

__global__ void gate_k(const float* __restrict__ xmean, const float* __restrict__ xmax,
                       const float* __restrict__ Weg, const float* __restrict__ beg,
                       float* __restrict__ xn, int total) {
  int i = blockIdx.x * blockDim.x + threadIdx.x;
  if (i >= total) return;
  int g = i >> 6, j = i & 63;
  float acc = beg[j];
  for (int c = 0; c < 32; ++c) acc += xmean[(size_t)g * 32 + c] * Weg[c * 64 + j];
  float gate = 1.0f / (1.0f + expf(-acc));
  float v = (j < 32) ? xmean[(size_t)g * 32 + j] : xmax[(size_t)g * 32 + (j - 32)];
  xn[i] = gate * v;
}

__global__ void head1_k(const float* __restrict__ xmean, const float* __restrict__ xn,
                        const float* __restrict__ Wf1, const float* __restrict__ bf1,
                        float* __restrict__ emb, int total) {
  int i = blockIdx.x * blockDim.x + threadIdx.x;
  if (i >= total) return;
  int g = i >> 5, o = i & 31;
  float acc = bf1[o];
  for (int k = 0; k < 32; ++k) acc += xmean[(size_t)g * 32 + k] * Wf1[k * 32 + o];
  for (int k = 0; k < 64; ++k) acc += xn[(size_t)g * 64 + k] * Wf1[(32 + k) * 32 + o];
  emb[i] = gelu_f(acc);
}

__global__ void head2_k(const float* __restrict__ emb, const float* __restrict__ Wf2,
                        const float* __restrict__ bf2, float* __restrict__ outp, int G) {
  int g = blockIdx.x * blockDim.x + threadIdx.x;
  if (g >= G) return;
  float acc = bf2[0];
  for (int c = 0; c < 32; ++c) acc += emb[(size_t)g * 32 + c] * Wf2[c];
  outp[g] = acc;
}

// ================================================================= launch
extern "C" void kernel_launch(void* const* d_in, const int* in_sizes, int n_in,
                              void* d_out, int out_size, void* d_ws, size_t ws_size,
                              hipStream_t stream) {
  (void)n_in; (void)out_size; (void)ws_size;
  const float* x     = (const float*)d_in[0];
  const int*   ei    = (const int*)d_in[1];
  const int*   batch = (const int*)d_in[2];
  const float* Wl1   = (const float*)d_in[3];
  const float* bl1   = (const float*)d_in[4];
  const float* Wr1   = (const float*)d_in[5];
  const float* br1   = (const float*)d_in[6];
  const float* att1  = (const float*)d_in[7];
  const float* bias1 = (const float*)d_in[8];
  const float* gn1w  = (const float*)d_in[9];
  const float* gn1b  = (const float*)d_in[10];
  const float* gn1ms = (const float*)d_in[11];
  const float* Wl2   = (const float*)d_in[12];
  const float* bl2   = (const float*)d_in[13];
  const float* Wr2   = (const float*)d_in[14];
  const float* br2   = (const float*)d_in[15];
  const float* att2  = (const float*)d_in[16];
  const float* bias2 = (const float*)d_in[17];
  const float* gn2w  = (const float*)d_in[18];
  const float* gn2b  = (const float*)d_in[19];
  const float* gn2ms = (const float*)d_in[20];
  const float* Wp1   = (const float*)d_in[21];
  const float* bp1   = (const float*)d_in[22];
  const float* Wp2   = (const float*)d_in[23];
  const float* bp2   = (const float*)d_in[24];
  const float* Weg   = (const float*)d_in[25];
  const float* beg   = (const float*)d_in[26];
  const float* Wf1   = (const float*)d_in[27];
  const float* bf1   = (const float*)d_in[28];
  const float* Wf2   = (const float*)d_in[29];
  const float* bf2   = (const float*)d_in[30];
  float* out = (float*)d_out;

  const int N  = in_sizes[0] / 128;
  const int E  = in_sizes[1] / 2;
  const int EL = E + N;
  const int G  = 512;

  size_t off = 0;
  auto carve = [&](size_t elems, size_t esz) -> void* {
    void* p = (char*)d_ws + off;
    off += (elems * esz + 255) & ~(size_t)255;
    return p;
  };
  _Float16* xbf = (_Float16*)carve((size_t)N * 128, 2);
  _Float16* hbf = (_Float16*)carve((size_t)N * 128, 2);
  _Float16* wbf = (_Float16*)carve(65536, 2);
  _Float16* wl1h = wbf;          _Float16* wr1h = wbf + 16384;
  _Float16* wp1h = wbf + 32768;  _Float16* wl2h = wbf + 49152;
  _Float16* wr2h = wbf + 53248;  _Float16* wp2h = wbf + 57344;

  float* xl1  = (float*)carve((size_t)N * 128, 4);
  float* xr1  = (float*)carve((size_t)N * 128, 4);
  float* res1 = (float*)carve((size_t)N * 128, 4);
  float* out1 = (float*)carve((size_t)N * 128, 4);
  float* h1   = (float*)carve((size_t)N * 128, 4);
  float* lg1  = (float*)carve((size_t)EL * 4, 4);
  float* m1   = (float*)carve((size_t)N * 4, 4);
  float* den1 = (float*)carve((size_t)N * 4, 4);
  float* xl2  = (float*)carve((size_t)N * 32, 4);
  float* xr2  = (float*)carve((size_t)N * 32, 4);
  float* res2 = (float*)carve((size_t)N * 32, 4);
  float* out2 = (float*)carve((size_t)N * 32, 4);
  float* h2   = (float*)carve((size_t)N * 32, 4);
  float* lg2  = (float*)carve((size_t)EL, 4);
  float* m2   = (float*)carve((size_t)N, 4);
  float* den2 = (float*)carve((size_t)N, 4);
  float* gns  = (float*)carve((size_t)G * 128, 4);
  float* gnv  = (float*)carve((size_t)G * 128, 4);
  float* gnr  = (float*)carve((size_t)G * 128, 4);
  float* cnt  = (float*)carve(G, 4);
  float* pm   = (float*)carve((size_t)G * 32, 4);
  float* px   = (float*)carve((size_t)G * 32, 4);
  float* xmean = (float*)carve((size_t)G * 32, 4);
  float* xn   = (float*)carve((size_t)G * 64, 4);
  float* emb  = (float*)carve((size_t)G * 32, 4);

  auto B = [](long long n) { return (unsigned)((n + THREADS - 1) / THREADS); };

  // ---- fp32 -> fp16 conversions (x and all GEMM weights)
  f2h_k<<<B((long long)N * 128), THREADS, 0, stream>>>(x, xbf, N * 128);
  f2h_k<<<B(16384), THREADS, 0, stream>>>(Wl1, wl1h, 16384);
  f2h_k<<<B(16384), THREADS, 0, stream>>>(Wr1, wr1h, 16384);
  f2h_k<<<B(16384), THREADS, 0, stream>>>(Wp1, wp1h, 16384);
  f2h_k<<<B(4096),  THREADS, 0, stream>>>(Wl2, wl2h, 4096);
  f2h_k<<<B(4096),  THREADS, 0, stream>>>(Wr2, wr2h, 4096);
  f2h_k<<<B(4096),  THREADS, 0, stream>>>(Wp2, wp2h, 4096);

  // ---- layer-1 GEMMs: [N,128] x [128,128] (NT=4 -> 2 column tiles)
  {
    int waves = ((N + 15) / 16) * 2;
    int blks = (waves + 7) / 8;
    gemm_wmma_k<4, 128><<<blks, THREADS, 0, stream>>>(xbf, wl1h, bl1, xl1, N, 128);
    gemm_wmma_k<4, 128><<<blks, THREADS, 0, stream>>>(xbf, wr1h, br1, xr1, N, 128);
    gemm_wmma_k<4, 128><<<blks, THREADS, 0, stream>>>(xbf, wp1h, bp1, res1, N, 128);
  }

  // ---- GATv2 conv1 (segment softmax over dst)
  fill_u32_k<<<B((long long)N * 4), THREADS, 0, stream>>>((unsigned*)m1, 0xFFFFFFFFu, N * 4);
  fill_u32_k<<<B((long long)N * 4), THREADS, 0, stream>>>((unsigned*)den1, 0u, N * 4);
  fill_u32_k<<<B((long long)N * 128), THREADS, 0, stream>>>((unsigned*)out1, 0u, N * 128);
  gat1_logit_k<<<(EL + 7) / 8, THREADS, 0, stream>>>(xl1, xr1, ei, att1, lg1, m1, E, EL);
  gat1_exp_k<<<B((long long)EL * 4), THREADS, 0, stream>>>(lg1, m1, den1, ei, E, EL);
  gat1_scatter_k<<<(EL + 7) / 8, THREADS, 0, stream>>>(xl1, lg1, den1, ei, out1, E, EL);
  addbias_k<128><<<B((long long)N * 128), THREADS, 0, stream>>>(out1, bias1, N * 128);

  // ---- GraphNorm1 + residual + gelu -> h1
  fill_u32_k<<<B(G * 128), THREADS, 0, stream>>>((unsigned*)gns, 0u, G * 128);
  fill_u32_k<<<B(G * 128), THREADS, 0, stream>>>((unsigned*)gnv, 0u, G * 128);
  fill_u32_k<<<B(G), THREADS, 0, stream>>>((unsigned*)cnt, 0u, G);
  count_k<<<B(N), THREADS, 0, stream>>>(batch, cnt, N);
  gn_sum_k<128><<<B((long long)N * 128), THREADS, 0, stream>>>(out1, batch, gns, N * 128);
  gn_mean_k<128><<<B(G * 128), THREADS, 0, stream>>>(gns, cnt, G * 128);
  gn_center_k<128><<<B((long long)N * 128), THREADS, 0, stream>>>(out1, batch, gns, gn1ms, gnv, N * 128);
  gn_rstd_k<128><<<B(G * 128), THREADS, 0, stream>>>(gnv, cnt, gnr, G * 128);
  gn_fin_k<128><<<B((long long)N * 128), THREADS, 0, stream>>>(out1, batch, gnr, gn1w, gn1b, res1, h1, N * 128);

  // ---- layer-2 GEMMs: [N,128] x [128,32] (NT=2 -> 1 column tile)
  f2h_k<<<B((long long)N * 128), THREADS, 0, stream>>>(h1, hbf, N * 128);
  {
    int waves = (N + 15) / 16;
    int blks = (waves + 7) / 8;
    gemm_wmma_k<2, 128><<<blks, THREADS, 0, stream>>>(hbf, wl2h, bl2, xl2, N, 32);
    gemm_wmma_k<2, 128><<<blks, THREADS, 0, stream>>>(hbf, wr2h, br2, xr2, N, 32);
    gemm_wmma_k<2, 128><<<blks, THREADS, 0, stream>>>(hbf, wp2h, bp2, res2, N, 32);
  }

  // ---- GATv2 conv2
  fill_u32_k<<<B(N), THREADS, 0, stream>>>((unsigned*)m2, 0xFFFFFFFFu, N);
  fill_u32_k<<<B(N), THREADS, 0, stream>>>((unsigned*)den2, 0u, N);
  fill_u32_k<<<B((long long)N * 32), THREADS, 0, stream>>>((unsigned*)out2, 0u, N * 32);
  gat2_logit_k<<<(EL + 7) / 8, THREADS, 0, stream>>>(xl2, xr2, ei, att2, lg2, m2, E, EL);
  gat2_exp_k<<<B(EL), THREADS, 0, stream>>>(lg2, m2, den2, ei, E, EL);
  gat2_scatter_k<<<(EL + 7) / 8, THREADS, 0, stream>>>(xl2, lg2, den2, ei, out2, E, EL);
  addbias_k<32><<<B((long long)N * 32), THREADS, 0, stream>>>(out2, bias2, N * 32);

  // ---- GraphNorm2 + residual + gelu -> h2 (reuse gns/gnv/gnr with D=32)
  fill_u32_k<<<B(G * 32), THREADS, 0, stream>>>((unsigned*)gns, 0u, G * 32);
  fill_u32_k<<<B(G * 32), THREADS, 0, stream>>>((unsigned*)gnv, 0u, G * 32);
  gn_sum_k<32><<<B((long long)N * 32), THREADS, 0, stream>>>(out2, batch, gns, N * 32);
  gn_mean_k<32><<<B(G * 32), THREADS, 0, stream>>>(gns, cnt, G * 32);
  gn_center_k<32><<<B((long long)N * 32), THREADS, 0, stream>>>(out2, batch, gns, gn2ms, gnv, N * 32);
  gn_rstd_k<32><<<B(G * 32), THREADS, 0, stream>>>(gnv, cnt, gnr, G * 32);
  gn_fin_k<32><<<B((long long)N * 32), THREADS, 0, stream>>>(out2, batch, gnr, gn2w, gn2b, res2, h2, N * 32);

  // ---- pooling (mean + max per graph) and ego-gated head
  fill_u32_k<<<B(G * 32), THREADS, 0, stream>>>((unsigned*)pm, 0u, G * 32);
  fill_u32_k<<<B(G * 32), THREADS, 0, stream>>>((unsigned*)px, 0xFFFFFFFFu, G * 32);
  pool_acc_k<<<B((long long)N * 32), THREADS, 0, stream>>>(h2, batch, pm, px, N * 32);
  pool_fin_k<<<B(G * 32), THREADS, 0, stream>>>(pm, px, cnt, xmean, G * 32);
  gate_k<<<B(G * 64), THREADS, 0, stream>>>(xmean, px, Weg, beg, xn, G * 64);
  head1_k<<<B(G * 32), THREADS, 0, stream>>>(xmean, xn, Wf1, bf1, emb, G * 32);
  head2_k<<<B(G), THREADS, 0, stream>>>(emb, Wf2, bf2, out, G);
}
